// Decoder_27882927685822
// MI455X (gfx1250) — compile-verified, matrix operand order checked by reading, not run on hardware
//
#include <hip/hip_runtime.h>

#define H_STEPS 8640   // 30*24*12
#define HID     128
#define BATCH   16
#define G4      512    // 4*HID

typedef __attribute__((ext_vector_type(16))) __bf16 v16bf;
typedef __attribute__((ext_vector_type(8)))  float  v8f;

union FragU {
    uint4          q[2];
    unsigned short u[16];
    v16bf          v;
};

// ---------------- LDS layout (bytes) ----------------
#define OFF_WH0   0         // 512*128 bf16 swizzled B-fragments  (131072)
#define OFF_WI1   131072    // 512*128 bf16 swizzled B-fragments  (131072)
#define OFF_G     262144    // 16*512 f32 gate buffer             (32768)
#define OFF_H0SWZ 294912    // h0 as bf16 A-fragments             (4096)
#define OFF_H1SWZ 299008    // h1 as bf16 A-fragments             (4096)
#define OFF_B0    303104    // combined bias layer0, 512 f32      (2048)
#define OFF_B1    305152    // combined bias layer1, 512 f32      (2048)
#define OFF_WI0   307200    // W_ih0 column, 512 f32              (2048)
#define OFF_FCW   309248    // fc weights, 128 f32                (512)
#define OFF_YBUF  309760    // double-buffered y, 2*16 f32        (128)
#define OFF_YPART 309888    // per-wave fc partials, 64 f32       (256)
#define SMEM_BYTES 310144   // <= 320KB WGP LDS

// float -> bf16 (round to nearest even)
__device__ __forceinline__ unsigned short f2bf(float f) {
    unsigned int u = __float_as_uint(f);
    u += 0x7FFFu + ((u >> 16) & 1u);
    return (unsigned short)(u >> 16);
}

// ISA 7.12.2 16-bit A-matrix fragment index: lane holds row m, K packed as
// k = 32*kc + 2*(v&3) + 16*(v>>2) + 8*half  (half = lane>>4)
__device__ __forceinline__ int swz(int m, int k) {
    int kc   = k >> 5;
    int kk   = k & 31;
    int v    = ((kk >> 4) << 2) | ((kk >> 1) & 3);
    int half = (kk >> 3) & 1;
    int lane = (half << 4) | m;
    return ((kc * 32 + lane) << 4) | ((v << 1) | (kk & 1));
}

__device__ __forceinline__ v16bf ldFrag(const char* p) {
    FragU f;
    const uint4* q = (const uint4*)p;
    f.q[0] = q[0];
    f.q[1] = q[1];
    return f.v;
}

__device__ __forceinline__ float sigm(float x) {
    return __fdividef(1.0f, 1.0f + __expf(-x));
}
__device__ __forceinline__ float tanh_(float x) {
    return __fdividef(2.0f, 1.0f + __expf(-2.0f * x)) - 1.0f;
}

__global__ __launch_bounds__(1024, 1)
void lstm_persistent(const float* __restrict__ y0,
                     const float* __restrict__ h0in,
                     const float* __restrict__ c0in,
                     const float* __restrict__ Wih0,
                     const float* __restrict__ Whh0,
                     const float* __restrict__ bih0,
                     const float* __restrict__ bhh0,
                     const float* __restrict__ Wih1,
                     const float* __restrict__ Whh1,
                     const float* __restrict__ bih1,
                     const float* __restrict__ bhh1,
                     const float* __restrict__ fcw,
                     const float* __restrict__ fcb,
                     float* __restrict__ out)
{
    extern __shared__ char smem[];
    unsigned short* wh0s  = (unsigned short*)(smem + OFF_WH0);
    unsigned short* wi1s  = (unsigned short*)(smem + OFF_WI1);
    float*          GL    = (float*)(smem + OFF_G);
    unsigned short* h0sL  = (unsigned short*)(smem + OFF_H0SWZ);
    unsigned short* h1sL  = (unsigned short*)(smem + OFF_H1SWZ);
    float*          b0L   = (float*)(smem + OFF_B0);
    float*          b1L   = (float*)(smem + OFF_B1);
    float*          wi0L  = (float*)(smem + OFF_WI0);
    float*          fcwL  = (float*)(smem + OFF_FCW);
    float*          ybufL = (float*)(smem + OFF_YBUF);
    float*          ypartL= (float*)(smem + OFF_YPART);

    const int tid  = threadIdx.x;
    const int w    = tid >> 5;   // wave id == output tile id (0..31)
    const int lane = tid & 31;

    // ---------------- one-time init ----------------
    // Swizzle Wh0 / Wi1 (f32 -> bf16 B-fragments) into LDS.
    for (int f = tid; f < 512 * 128; f += 1024) {
        int wt = f >> 11, r = f & 2047;
        int kc = r >> 9, ln = (r >> 4) & 31, vv = (r >> 1) & 7, p = r & 1;
        int n = ln & 15, half = ln >> 4;
        int k   = kc * 32 + 2 * (vv & 3) + 16 * (vv >> 2) + 8 * half + p;
        int row = wt * 16 + n;
        wh0s[f] = f2bf(Whh0[row * HID + k]);
        wi1s[f] = f2bf(Wih1[row * HID + k]);
    }
    for (int c = tid; c < 512; c += 1024) {
        b0L[c]  = bih0[c] + bhh0[c];
        b1L[c]  = bih1[c] + bhh1[c];
        wi0L[c] = Wih0[c];
    }
    for (int c = tid; c < 128; c += 1024) fcwL[c] = fcw[c];
    if (tid < 16) ybufL[tid] = y0[tid];

    // Cell state lives in registers: each thread owns 2 elements per layer.
    float c0r[2], c1r[2];
#pragma unroll
    for (int idx = 0; idx < 2; ++idx) {
        int e = tid + (idx << 10);
        int b = e >> 7, j = e & 127;
        c0r[idx] = c0in[b * HID + j];
        c1r[idx] = c0in[2048 + b * HID + j];
        h0sL[swz(b, j)] = f2bf(h0in[b * HID + j]);
        h1sL[swz(b, j)] = f2bf(h0in[2048 + b * HID + j]);
    }

    // Wh1 tile lives persistently in VGPRs (4 x v16bf per lane = 32 VGPRs).
    v16bf wh1f[4];
    {
        int n = lane & 15, half = lane >> 4;
        for (int kc = 0; kc < 4; ++kc) {
            FragU fr;
#pragma unroll
            for (int vv = 0; vv < 8; ++vv)
#pragma unroll
                for (int p = 0; p < 2; ++p) {
                    int k = kc * 32 + 2 * (vv & 3) + 16 * (vv >> 2) + 8 * half + p;
                    fr.u[vv * 2 + p] = f2bf(Whh1[(w * 16 + n) * HID + k]);
                }
            wh1f[kc] = fr.v;
        }
    }
    const float fcbv = fcb[0];
    __syncthreads();

    // ---------------- persistent time loop ----------------
    for (int t = 0; t < H_STEPS; ++t) {
        const float* ycur = ybufL + ((t & 1) << 4);
        float*       ynxt = ybufL + (((t + 1) & 1) << 4);

        // Phase A: G0 = h0 @ Wh0^T  (one 16x16 tile per wave)
        {
            v8f acc = {};
#pragma unroll
            for (int kc = 0; kc < 4; ++kc) {
                v16bf a = ldFrag(smem + OFF_H0SWZ + (((kc << 5) | lane) << 5));
                v16bf b = ldFrag(smem + OFF_WH0 + ((((w << 2) | kc) * 32 + lane) << 5));
                acc = __builtin_amdgcn_wmma_f32_16x16x32_bf16(
                    false, a, false, b, (short)0, acc, false, false);
            }
            int col = (w << 4) | (lane & 15);
            int rb  = (lane >> 4) << 3;
#pragma unroll
            for (int r = 0; r < 8; ++r) GL[(rb + r) * G4 + col] = acc[r];
        }
        __syncthreads();

        // Phase B: layer-0 activations; c0 in regs; h0 -> bf16 A-fragments
#pragma unroll
        for (int idx = 0; idx < 2; ++idx) {
            int e = tid + (idx << 10);
            int b = e >> 7, j = e & 127;
            float yb = ycur[b];
            const float* Gr = GL + b * G4;
            float xi = Gr[j]       + b0L[j]       + yb * wi0L[j];
            float xf = Gr[128 + j] + b0L[128 + j] + yb * wi0L[128 + j];
            float xg = Gr[256 + j] + b0L[256 + j] + yb * wi0L[256 + j];
            float xo = Gr[384 + j] + b0L[384 + j] + yb * wi0L[384 + j];
            float cc = sigm(xf) * c0r[idx] + sigm(xi) * tanh_(xg);
            c0r[idx] = cc;
            float hh = sigm(xo) * tanh_(cc);
            h0sL[swz(b, j)] = f2bf(hh);
        }
        __syncthreads();

        // Phase C: G1 = h0n @ Wi1^T + h1 @ Wh1^T  (Wh1 from registers)
        {
            v8f acc = {};
#pragma unroll
            for (int kc = 0; kc < 4; ++kc) {
                v16bf a = ldFrag(smem + OFF_H0SWZ + (((kc << 5) | lane) << 5));
                v16bf b = ldFrag(smem + OFF_WI1 + ((((w << 2) | kc) * 32 + lane) << 5));
                acc = __builtin_amdgcn_wmma_f32_16x16x32_bf16(
                    false, a, false, b, (short)0, acc, false, false);
            }
#pragma unroll
            for (int kc = 0; kc < 4; ++kc) {
                v16bf a = ldFrag(smem + OFF_H1SWZ + (((kc << 5) | lane) << 5));
                acc = __builtin_amdgcn_wmma_f32_16x16x32_bf16(
                    false, a, false, wh1f[kc], (short)0, acc, false, false);
            }
            int col = (w << 4) | (lane & 15);
            int rb  = (lane >> 4) << 3;
#pragma unroll
            for (int r = 0; r < 8; ++r) GL[(rb + r) * G4 + col] = acc[r];
        }
        __syncthreads();

        // Phase D: layer-1 activations; c1 in regs; h1 fragments; fc partials
        float part0, part1;
        {
            float parts[2];
#pragma unroll
            for (int idx = 0; idx < 2; ++idx) {
                int e = tid + (idx << 10);
                int b = e >> 7, j = e & 127;
                const float* Gr = GL + b * G4;
                float xi = Gr[j]       + b1L[j];
                float xf = Gr[128 + j] + b1L[128 + j];
                float xg = Gr[256 + j] + b1L[256 + j];
                float xo = Gr[384 + j] + b1L[384 + j];
                float cc = sigm(xf) * c1r[idx] + sigm(xi) * tanh_(xg);
                c1r[idx] = cc;
                float hh = sigm(xo) * tanh_(cc);
                h1sL[swz(b, j)] = f2bf(hh);
                parts[idx] = hh * fcwL[j];
            }
            part0 = parts[0];
            part1 = parts[1];
        }
        // all 32 lanes of a wave share the same batch index -> wave reduce
#pragma unroll
        for (int off = 16; off >= 1; off >>= 1) {
            part0 += __shfl_xor(part0, off, 32);
            part1 += __shfl_xor(part1, off, 32);
        }
        if (lane == 0) {
            ypartL[w]      = part0;   // batch  w>>2
            ypartL[32 + w] = part1;   // batch (w>>2)+8
        }
        __syncthreads();

        // finalize y (deterministic 4-way sum), emit output, feed next step
        if (tid < 16) {
            int base = (tid < 8) ? (tid << 2) : (32 + ((tid - 8) << 2));
            float s = fcbv + ypartL[base] + ypartL[base + 1] +
                      ypartL[base + 2] + ypartL[base + 3];
            ynxt[tid] = s;
            out[tid * H_STEPS + t] = s;
        }
    }
}

extern "C" void kernel_launch(void* const* d_in, const int* in_sizes, int n_in,
                              void* d_out, int out_size, void* d_ws, size_t ws_size,
                              hipStream_t stream) {
    (void)in_sizes; (void)n_in; (void)d_ws; (void)ws_size; (void)out_size;
    const float* y0   = (const float*)d_in[0];
    const float* h0   = (const float*)d_in[1];
    const float* c0   = (const float*)d_in[2];
    const float* Wih0 = (const float*)d_in[3];
    const float* Whh0 = (const float*)d_in[4];
    const float* bih0 = (const float*)d_in[5];
    const float* bhh0 = (const float*)d_in[6];
    const float* Wih1 = (const float*)d_in[7];
    const float* Whh1 = (const float*)d_in[8];
    const float* bih1 = (const float*)d_in[9];
    const float* bhh1 = (const float*)d_in[10];
    const float* fcw  = (const float*)d_in[11];
    const float* fcb  = (const float*)d_in[12];
    float* out = (float*)d_out;

    (void)hipFuncSetAttribute((const void*)lstm_persistent,
                              hipFuncAttributeMaxDynamicSharedMemorySize,
                              SMEM_BYTES);

    lstm_persistent<<<dim3(1), dim3(1024), SMEM_BYTES, stream>>>(
        y0, h0, c0, Wih0, Whh0, bih0, bhh0, Wih1, Whh1, bih1, bhh1,
        fcw, fcb, out);
}
